// KLayerGAT_30133490549159
// MI455X (gfx1250) — matrix operand order repaired and checked by reading, not compiled, not run on hardware
//
#include <hip/hip_runtime.h>

// GATv2 3-layer forward for MI455X (gfx1250, wave32).
// GEMMs via v_wmma_f32_16x16x32_bf16; edge phase L2-resident with f32 atomics.

#define NNODES 50000
#define NEDGES 800000
#define HEADS  2

typedef __attribute__((ext_vector_type(16))) __bf16 v16bf;
typedef __attribute__((ext_vector_type(8)))  __bf16 v8bf;
typedef __attribute__((ext_vector_type(8)))  float  v8f;

// ---- monotone float<->uint key for atomicMax-based segment max ----
__device__ __forceinline__ unsigned f32_key(float f) {
  unsigned b = __float_as_uint(f);
  return (b & 0x80000000u) ? ~b : (b | 0x80000000u);
}
__device__ __forceinline__ float key_f32(unsigned k) {
  unsigned b = (k & 0x80000000u) ? (k & 0x7fffffffu) : ~k;
  return __uint_as_float(b);
}

// ---------------- utility kernels ----------------
__global__ void k_fill_f32(float* p, float v, int n) {
  int i = blockIdx.x * blockDim.x + threadIdx.x;
  if (i < n) p[i] = v;
}
__global__ void k_fill_u32(unsigned* p, unsigned v, int n) {
  int i = blockIdx.x * blockDim.x + threadIdx.x;
  if (i < n) p[i] = v;
}

// features f32 -> bf16, optional ELU (for layer outputs feeding next layer)
__global__ void k_feat_to_bf16(const float* __restrict__ in, __bf16* __restrict__ out,
                               int n, int applyElu) {
  int i = blockIdx.x * blockDim.x + threadIdx.x;
  if (i >= n) return;
  float v = in[i];
  if (applyElu) v = v > 0.f ? v : expm1f(v);
  out[i] = (__bf16)v;
}

// W (256 x dout, row-major f32) -> Wt (dout x 256, row-major bf16)
__global__ void k_weight_to_bf16_t(const float* __restrict__ W, __bf16* __restrict__ Wt,
                                   int dout) {
  int i = blockIdx.x * blockDim.x + threadIdx.x;
  int total = 256 * dout;
  if (i >= total) return;
  int k = i / dout, n = i - k * dout;
  Wt[n * 256 + k] = (__bf16)W[i];
}

// ---------------- WMMA GEMM: C(N x dout) = A(N x 256) * W + bias ----------------
// A: row-major bf16 (row stride 256). Bt: W^T, row-major bf16 (row stride 256).
// One wave per 16x16 output tile; K looped 8x32.
__global__ void __launch_bounds__(256)
k_gemm_wmma_bf16(const __bf16* __restrict__ A, const __bf16* __restrict__ Bt,
                 const float* __restrict__ bias, float* __restrict__ C, int dout) {
  const int wave = (blockIdx.x * blockDim.x + threadIdx.x) >> 5;
  const int lane = threadIdx.x & 31;
  const int ntiles = dout >> 4;
  const int tm = wave / ntiles;
  const int tn = wave - tm * ntiles;
  if (tm * 16 >= NNODES) return;              // wave-uniform guard (EXEC all-1 for WMMA)
  const int half = lane >> 4;
  const int l15  = lane & 15;
  const __bf16* arow = A  + (tm * 16 + l15) * 256;   // A fragment: M = lane&15
  const __bf16* brow = Bt + (tn * 16 + l15) * 256;   // B fragment: N = lane&15
  // A (16-bit 16x32): lanes 0-15 hold K {0..7,16..23}; lanes 16-31 hold {8..15,24..31}
  const int ka0 = half ? 8  : 0;
  const int ka1 = half ? 24 : 16;
  // B (16-bit 32x16): lanes 0-15 hold K {0..7,16..23}; lanes 16-31 hold {8..15,24..31}
  const int kb0 = half ? 8 : 0;
  v8f acc = {};
  #pragma unroll
  for (int kb = 0; kb < 256; kb += 32) {
    v8bf a0 = *(const v8bf*)(arow + kb + ka0);
    v8bf a1 = *(const v8bf*)(arow + kb + ka1);
    v8bf b0 = *(const v8bf*)(brow + kb + kb0);
    v8bf b1 = *(const v8bf*)(brow + kb + kb0 + 16);
    v16bf av, bv;
    #pragma unroll
    for (int i = 0; i < 8; ++i) {
      av[i] = a0[i]; av[i + 8] = a1[i];
      bv[i] = b0[i]; bv[i + 8] = b1[i];
    }
    acc = __builtin_amdgcn_wmma_f32_16x16x32_bf16(false, av, false, bv,
                                                  (short)0, acc, false, false);
  }
  // C/D layout: VGPR g -> M = g (+8 for upper lane half), N = lane&15
  const int col = tn * 16 + l15;
  const float bb = bias[col];
  #pragma unroll
  for (int g = 0; g < 8; ++g) {
    int row = tm * 16 + g + (half ? 8 : 0);
    C[row * dout + col] = acc[g] + bb;
  }
}

// ---------------- edge kernels (wave per (edge, head)) ----------------
// logits[e,h] = attn[h] . leaky_relu(el[src] + er[dst]); also atomic segment-max
__global__ void __launch_bounds__(256)
k_edge_logits(const int* __restrict__ src, const int* __restrict__ dst,
              const float* __restrict__ el, const float* __restrict__ er,
              const float* __restrict__ attn, float* __restrict__ lg,
              unsigned* __restrict__ mkey, int fo) {
  const int w = (blockIdx.x * blockDim.x + threadIdx.x) >> 5;
  if (w >= NEDGES * HEADS) return;
  const int lane = threadIdx.x & 31;
  const int e = w >> 1, head = w & 1;
  const int s = src[e], d = dst[e];
  const int dout = fo * HEADS;
  const int fpl  = fo >> 5;                 // features per lane (4 or 2)
  const int base = head * fo + lane * fpl;
  float p = 0.f;
  for (int j = 0; j < fpl; j += 2) {
    float2 lv = *(const float2*)(el + s * dout + base + j);
    float2 rv = *(const float2*)(er + d * dout + base + j);
    float2 av = *(const float2*)(attn + head * fo + lane * fpl + j);
    float v0 = lv.x + rv.x; v0 = v0 > 0.f ? v0 : 0.2f * v0;
    float v1 = lv.y + rv.y; v1 = v1 > 0.f ? v1 : 0.2f * v1;
    p += v0 * av.x + v1 * av.y;
  }
  for (int off = 16; off; off >>= 1) p += __shfl_xor(p, off, 32);
  if (lane == 0) {
    lg[w] = p;
    atomicMax(mkey + d * HEADS + head, f32_key(p));
  }
}

// ex = exp(logit - m[dst]); overwrite lg with ex; atomic segment-sum
__global__ void k_edge_expsum(const int* __restrict__ dst, float* __restrict__ lg,
                              const unsigned* __restrict__ mkey, float* __restrict__ ssum) {
  int i = blockIdx.x * blockDim.x + threadIdx.x;
  if (i >= NEDGES * HEADS) return;
  int e = i >> 1, head = i & 1;
  int d = dst[e];
  float m = key_f32(mkey[d * HEADS + head]);
  float ex = __expf(lg[i] - m);
  lg[i] = ex;
  atomicAdd(ssum + d * HEADS + head, ex);
}

// agg[dst] += (ex / s[dst]) * el[src]
__global__ void __launch_bounds__(256)
k_edge_aggregate(const int* __restrict__ src, const int* __restrict__ dst,
                 const float* __restrict__ el, const float* __restrict__ lg,
                 const float* __restrict__ ssum, float* __restrict__ agg, int fo) {
  const int w = (blockIdx.x * blockDim.x + threadIdx.x) >> 5;
  if (w >= NEDGES * HEADS) return;
  const int lane = threadIdx.x & 31;
  const int e = w >> 1, head = w & 1;
  const int s = src[e], d = dst[e];
  const int dout = fo * HEADS;
  const int fpl  = fo >> 5;
  const int base = head * fo + lane * fpl;
  const float a = lg[w] / ssum[d * HEADS + head];
  for (int j = 0; j < fpl; j += 2) {
    float2 v = *(const float2*)(el + s * dout + base + j);
    atomicAdd(agg + d * dout + base + j,     a * v.x);
    atomicAdd(agg + d * dout + base + j + 1, a * v.y);
  }
}

// final head-mean: out[n, c] = 0.5 * (agg[n, 0, c] + agg[n, 1, c]), NC = 64
__global__ void k_head_mean(const float* __restrict__ agg, float* __restrict__ out) {
  int i = blockIdx.x * blockDim.x + threadIdx.x;
  if (i >= NNODES * 64) return;
  int n = i >> 6, c = i & 63;
  out[i] = 0.5f * (agg[n * 128 + c] + agg[n * 128 + 64 + c]);
}

// ---------------- launcher ----------------
extern "C" void kernel_launch(void* const* d_in, const int* in_sizes, int n_in,
                              void* d_out, int out_size, void* d_ws, size_t ws_size,
                              hipStream_t stream) {
  (void)in_sizes; (void)n_in; (void)out_size; (void)ws_size;
  const float* x   = (const float*)d_in[0];
  const int*   src = (const int*)d_in[1];
  const int*   dst = (const int*)d_in[2];
  const float* Ws[3] = {(const float*)d_in[3], (const float*)d_in[8],  (const float*)d_in[13]};
  const float* bs[3] = {(const float*)d_in[4], (const float*)d_in[9],  (const float*)d_in[14]};
  const float* Wd[3] = {(const float*)d_in[5], (const float*)d_in[10], (const float*)d_in[15]};
  const float* bd[3] = {(const float*)d_in[6], (const float*)d_in[11], (const float*)d_in[16]};
  const float* at[3] = {(const float*)d_in[7], (const float*)d_in[12], (const float*)d_in[17]};

  // workspace layout (bytes, all offsets 16B-aligned)
  char* ws = (char*)d_ws;
  __bf16*   hb   = (__bf16*)(ws);                 // N*256 bf16   = 25,600,000
  __bf16*   wsb  = (__bf16*)(ws +  25600000);     // 256*256 bf16 =    131,072
  __bf16*   wdb  = (__bf16*)(ws +  25731072);     // 256*256 bf16 =    131,072
  float*    el   = (float*)  (ws +  25862144);    // N*256 f32    = 51,200,000
  float*    er   = (float*)  (ws +  77062144);    // N*256 f32    = 51,200,000
  float*    lg   = (float*)  (ws + 128262144);    // E*2 f32      =  6,400,000
  unsigned* mk   = (unsigned*)(ws + 134662144);   // N*2 u32      =    400,000
  float*    ssum = (float*)  (ws + 135062144);    // N*2 f32      =    400,000
  float*    agg  = (float*)  (ws + 135462144);    // N*256 f32    = 51,200,000
                                                  // total ~186.7 MB

  const int fos[3] = {128, 128, 64};              // out feats per head

  for (int li = 0; li < 3; ++li) {
    const int fo = fos[li], dout = fo * HEADS;
    // 1) features -> bf16 (ELU on previous layer's agg for li>0); must precede agg clear
    {
      int n = NNODES * 256;
      k_feat_to_bf16<<<(n + 255) / 256, 256, 0, stream>>>(li == 0 ? x : agg, hb, n, li != 0);
    }
    // 2) weights -> transposed bf16
    {
      int n = 256 * dout;
      k_weight_to_bf16_t<<<(n + 255) / 256, 256, 0, stream>>>(Ws[li], wsb, dout);
      k_weight_to_bf16_t<<<(n + 255) / 256, 256, 0, stream>>>(Wd[li], wdb, dout);
    }
    // 3) clear accumulators (every call: harness does not re-initialize ws)
    k_fill_f32<<<(NNODES * dout + 255) / 256, 256, 0, stream>>>(agg, 0.f, NNODES * dout);
    k_fill_f32<<<(NNODES * HEADS + 255) / 256, 256, 0, stream>>>(ssum, 0.f, NNODES * HEADS);
    k_fill_u32<<<(NNODES * HEADS + 255) / 256, 256, 0, stream>>>(mk, 0u, NNODES * HEADS);
    // 4) dense GEMMs via WMMA: el = hb@Ws + bs ; er = hb@Wd + bd
    {
      int waves  = (NNODES / 16) * (dout / 16);
      int blocks = (waves + 7) / 8;               // 8 waves (256 thr) per block
      k_gemm_wmma_bf16<<<blocks, 256, 0, stream>>>(hb, wsb, bs[li], el, dout);
      k_gemm_wmma_bf16<<<blocks, 256, 0, stream>>>(hb, wdb, bd[li], er, dout);
    }
    // 5) edge softmax-attention + scatter aggregation
    {
      int waves = NEDGES * HEADS;
      k_edge_logits   <<<(waves + 7) / 8,     256, 0, stream>>>(src, dst, el, er, at[li], lg, mk, fo);
      k_edge_expsum   <<<(waves + 255) / 256, 256, 0, stream>>>(dst, lg, mk, ssum);
      k_edge_aggregate<<<(waves + 7) / 8,     256, 0, stream>>>(src, dst, el, lg, ssum, agg, fo);
    }
  }
  // 6) mean over heads -> (N, 64) output
  k_head_mean<<<(NNODES * 64 + 255) / 256, 256, 0, stream>>>(agg, (float*)d_out);
}
